// InteractionMaskBlock_9517647528609
// MI455X (gfx1250) — matrix operand order repaired and checked by reading, not compiled, not run on hardware
//
#include <hip/hip_runtime.h>
#include <hip/hip_bf16.h>
#include <math.h>

// ---------------- problem constants ----------------
#define DIMC 768
#define NHH  8
#define NT   197          // tokens
#define NTP  224          // padded tokens (14*16 = 7*32)
#define HD   96           // head dim
#define BB   64
#define MROWS (BB * NT)   // 12608 = 788*16 = 394*32
#define HIDC 3072
#define QKVC 2304

// ---------------- vector types ----------------
typedef __attribute__((ext_vector_type(16))) __bf16 v16bf;
typedef __attribute__((ext_vector_type(8)))  float  v8f;
typedef __attribute__((ext_vector_type(4)))  float  f32x4;
typedef __attribute__((ext_vector_type(4)))  unsigned int u32x4;
typedef __attribute__((ext_vector_type(8)))  int    i32x8;
typedef __attribute__((ext_vector_type(4)))  int    i32x4;

union FragU { v16bf v; f32x4 q[2]; };

#if defined(__has_builtin)
# if __has_builtin(__builtin_amdgcn_tensor_load_to_lds)
#  define HAVE_TDM 1
# endif
#endif

// A fragment (16x32, MxK): lane L -> row L&15; elems 0..7 at K=kb+(L>>4)*8, elems 8..15 at +16.
__device__ __forceinline__ v16bf load_a_ptr(const __bf16* p) {
  FragU u;
  u.q[0] = *reinterpret_cast<const f32x4*>(p);
  u.q[1] = *reinterpret_cast<const f32x4*>(p + 16);
  return u.v;
}
// B fragment (32x16, KxN) from K-major storage: lane L -> column L&15; elems 0..7 at
// K=kb+(L>>4)*8, elems 8..15 at +16 (stride ld).
__device__ __forceinline__ v16bf load_b_ptr(const __bf16* p, int ld) {
  v16bf v;
#pragma unroll
  for (int s = 0; s < 8; ++s) { v[s] = p[(size_t)s * ld]; v[s + 8] = p[(size_t)(s + 16) * ld]; }
  return v;
}
__device__ __forceinline__ v16bf load_b_frag(const __bf16* base, int ldb, int kb, int lane) {
  return load_b_ptr(base + (size_t)(kb + ((lane >> 4) << 3)) * ldb + (lane & 15), ldb);
}

#define WMMA_BF16(a, b, c) \
  __builtin_amdgcn_wmma_f32_16x16x32_bf16(false, (a), false, (b), (short)0, (c), false, false)

// LDS B-panel geometry: 64 rows x 128 cols bf16, 8-element (16B) row pad -> 136-elem stride
#define BROWS 64
#define BCOLS 128
#define BSTRIDE 136

#ifdef HAVE_TDM
// Issue a TDM 2D tile load: 64x128 bf16 tile at gsrc (row stride N elems) -> LDS at lds_off,
// with LDS padding of 4 DWORDs every 64 DWORDs (matches BSTRIDE).
__device__ __forceinline__ void tdm_load_b_tile(const __bf16* gsrc, unsigned lds_off, int N) {
  unsigned long long ga = (unsigned long long)(size_t)gsrc;
  u32x4 g0;
  g0.x = 1u;                                            // count=1 (valid), user mode
  g0.y = lds_off;                                       // LDS byte address
  g0.z = (unsigned)(ga & 0xFFFFFFFFull);                // global addr [31:0]
  g0.w = (unsigned)((ga >> 32) & 0x1FFFFFFu) | (2u << 30); // addr[56:32], type=2 (image)
  i32x8 g1;
  g1[0] = (1 << 16) | (1 << 20) | (5 << 22) | (3 << 25); // data_size=2B, pad_en, 64DW interval, 4DW pad
  g1[1] = (int)((unsigned)(BCOLS & 0xFFFF) << 16);       // tensor_dim0 lo16 (==tile_dim0: no OOB)
  g1[2] = (int)(((unsigned)BCOLS >> 16) | ((unsigned)(BROWS & 0xFFFF) << 16)); // dim0 hi | dim1 lo
  g1[3] = (int)(((unsigned)BROWS >> 16) | ((unsigned)BCOLS << 16));            // dim1 hi | tile_dim0
  g1[4] = BROWS;                                         // tile_dim1=64, tile_dim2=0
  g1[5] = N;                                             // tensor_dim0_stride low32 (elements)
  g1[6] = 0;                                             // stride0 hi16 | stride1 lo16
  g1[7] = 0;                                             // stride1 hi32
  i32x4 z4 = {0, 0, 0, 0};
#if __clang_major__ >= 23
  i32x8 z8 = {0, 0, 0, 0, 0, 0, 0, 0};
  __builtin_amdgcn_tensor_load_to_lds(g0, g1, z4, z4, z8, 0);
#else
  __builtin_amdgcn_tensor_load_to_lds(g0, g1, z4, z4, 0);
#endif
}
#endif

// ---------------- weight convert + transpose: src (Nout,K) fp32 -> dst (K,Nout) bf16
__global__ void convT_kernel(const float* __restrict__ src, __bf16* __restrict__ dst,
                             int Nout, int K) {
  size_t idx = (size_t)blockIdx.x * 256 + threadIdx.x;
  if (idx >= (size_t)Nout * K) return;
  int k = (int)(idx % K);
  int n = (int)(idx / K);
  dst[(size_t)k * Nout + n] = (__bf16)src[idx];
}

// ---------------- LayerNorm over 768, fp32 in -> bf16 out (one block per row)
__global__ void ln_bf16_kernel(const float* __restrict__ x, const float* __restrict__ w,
                               const float* __restrict__ b, __bf16* __restrict__ out) {
  int row = blockIdx.x, tid = threadIdx.x;
  const float* xr = x + (size_t)row * DIMC;
  __shared__ float red[8];
  __shared__ float stat[2];
  float vals[3];
  float s = 0.f;
#pragma unroll
  for (int e = 0; e < 3; ++e) { vals[e] = xr[tid + 256 * e]; s += vals[e]; }
  for (int m = 16; m; m >>= 1) s += __shfl_xor(s, m, 32);
  if ((tid & 31) == 0) red[tid >> 5] = s;
  __syncthreads();
  if (tid == 0) { float t = 0.f; for (int i = 0; i < 8; ++i) t += red[i]; stat[0] = t / (float)DIMC; }
  __syncthreads();
  float mean = stat[0];
  float sv = 0.f;
#pragma unroll
  for (int e = 0; e < 3; ++e) { float d = vals[e] - mean; sv += d * d; }
  for (int m = 16; m; m >>= 1) sv += __shfl_xor(sv, m, 32);
  __syncthreads();
  if ((tid & 31) == 0) red[tid >> 5] = sv;
  __syncthreads();
  if (tid == 0) { float t = 0.f; for (int i = 0; i < 8; ++i) t += red[i]; stat[1] = rsqrtf(t / (float)DIMC + 1e-5f); }
  __syncthreads();
  float inv = stat[1];
#pragma unroll
  for (int e = 0; e < 3; ++e) {
    int c = tid + 256 * e;
    out[(size_t)row * DIMC + c] = (__bf16)((vals[e] - mean) * inv * w[c] + b[c]);
  }
}

// ---------------- WMMA GEMM with TDM-staged B panel in LDS ----------------
// out(MxN) = A(MxK)bf16 @ B(KxN)bf16 (B stored K-major). Block: 256 thr = 8 waves,
// 128x128 output tile; wave = 32 rows x 64 cols (2 A frags x 4 B frags x 8 acc).
// mode 0: bf16 (+bias)  1: GELU->bf16  2: fp32 = res+acc+bias  3: fp32 += acc+bias
__global__ void gemm_bf16_kernel(const __bf16* __restrict__ A, const __bf16* __restrict__ Bm,
                                 const float* __restrict__ bias, const float* __restrict__ res,
                                 float* __restrict__ outF, __bf16* __restrict__ outH,
                                 int M, int N, int K, int mode) {
  __shared__ __align__(16) __bf16 Bs[2][BROWS * BSTRIDE];
  int w = threadIdx.x >> 5, lane = threadIdx.x & 31;
  int rg = w & 3, cg = w >> 2;
  int rowA = blockIdx.y * 128 + rg * 32;
  int col0 = blockIdx.x * 128;
  bool live = rowA < M;                        // M % 32 == 0 for all our calls
  int rowBase = live ? rowA : 0;

  const __bf16* pA0 = A + (size_t)(rowBase + (lane & 15)) * K + ((lane >> 4) << 3);
  const __bf16* pA1 = pA0 + (size_t)16 * K;
  // per-lane LDS read base for B fragments (K row 8*(lane>>4), col lane&15 + cg*64)
  int bOff = ((lane >> 4) << 3) * BSTRIDE + (lane & 15) + cg * 64;

  v8f acc[2][4];
#pragma unroll
  for (int pr = 0; pr < 2; ++pr)
#pragma unroll
    for (int t = 0; t < 4; ++t)
#pragma unroll
      for (int i = 0; i < 8; ++i) acc[pr][t][i] = 0.f;

  int nc = K >> 6;                             // 64-row K chunks
#ifdef HAVE_TDM
  if (threadIdx.x < 32)
    tdm_load_b_tile(Bm + col0, (unsigned)(size_t)&Bs[0][0], N);
#endif
  for (int c = 0; c < nc; ++c) {
    __syncthreads();                           // prior reads of next buffer finished
#ifdef HAVE_TDM
    if (threadIdx.x < 32) {
      if (c + 1 < nc) {
        tdm_load_b_tile(Bm + (size_t)((c + 1) << 6) * N + col0,
                        (unsigned)(size_t)&Bs[(c + 1) & 1][0], N);
        __builtin_amdgcn_s_wait_tensorcnt(1);  // chunk c landed
      } else {
        __builtin_amdgcn_s_wait_tensorcnt(0);
      }
    }
#else
    // cooperative copy of chunk c: 64x128 bf16, 32 elems per thread
    {
      const __bf16* src = Bm + (size_t)(c << 6) * N + col0;
      int r = threadIdx.x >> 3;
      int cc = (threadIdx.x & 7) * 16;
      for (int rr = r; rr < BROWS; rr += 32) {
        f32x4 d0 = *reinterpret_cast<const f32x4*>(src + (size_t)rr * N + cc);
        *reinterpret_cast<f32x4*>(&Bs[c & 1][rr * BSTRIDE + cc]) = d0;
        f32x4 d1 = *reinterpret_cast<const f32x4*>(src + (size_t)rr * N + cc + 8);
        *reinterpret_cast<f32x4*>(&Bs[c & 1][rr * BSTRIDE + cc + 8]) = d1;
      }
    }
#endif
    __syncthreads();                           // chunk c visible to all waves
    const __bf16* bs = &Bs[c & 1][0] + bOff;
    __builtin_prefetch(pA0 + 128, 0, 0);       // global_prefetch_b8 for A stream
#pragma unroll
    for (int kk = 0; kk < 2; ++kk) {
      // preload ALL four B fragments + both A fragments into distinct registers,
      // then issue 8 independent-accumulator WMMAs back-to-back.
      v16bf b0 = load_b_ptr(bs + kk * 32 * BSTRIDE, BSTRIDE);
      v16bf b1 = load_b_ptr(bs + kk * 32 * BSTRIDE + 16, BSTRIDE);
      v16bf b2 = load_b_ptr(bs + kk * 32 * BSTRIDE + 32, BSTRIDE);
      v16bf b3 = load_b_ptr(bs + kk * 32 * BSTRIDE + 48, BSTRIDE);
      v16bf a0 = load_a_ptr(pA0);
      v16bf a1 = load_a_ptr(pA1);
      pA0 += 32; pA1 += 32;
      acc[0][0] = WMMA_BF16(a0, b0, acc[0][0]);
      acc[1][0] = WMMA_BF16(a1, b0, acc[1][0]);
      acc[0][1] = WMMA_BF16(a0, b1, acc[0][1]);
      acc[1][1] = WMMA_BF16(a1, b1, acc[1][1]);
      acc[0][2] = WMMA_BF16(a0, b2, acc[0][2]);
      acc[1][2] = WMMA_BF16(a1, b2, acc[1][2]);
      acc[0][3] = WMMA_BF16(a0, b3, acc[0][3]);
      acc[1][3] = WMMA_BF16(a1, b3, acc[1][3]);
    }
  }

  if (!live) return;
  int r = (lane >> 4) << 3, cc = lane & 15;
#pragma unroll
  for (int pr = 0; pr < 2; ++pr) {
#pragma unroll
    for (int i2 = 0; i2 < 8; ++i2) {
      int row = rowA + pr * 16 + r + i2;
#pragma unroll
      for (int t = 0; t < 4; ++t) {
        int col = col0 + cg * 64 + t * 16 + cc;
        float v = acc[pr][t][i2] + (bias ? bias[col] : 0.f);
        size_t idx = (size_t)row * N + col;
        if (mode == 0) {
          outH[idx] = (__bf16)v;
        } else if (mode == 1) {
          v = 0.5f * v * (1.f + erff(v * 0.70710678118654752f));
          outH[idx] = (__bf16)v;
        } else if (mode == 2) {
          outF[idx] = res[idx] + v;
        } else {
          outF[idx] += v;
        }
      }
    }
  }
}

// ---------------- QKV post: l2-normalize q,k per head, fold min(scale,100) into q ----------------
__global__ void qkv_post_kernel(const __bf16* __restrict__ QKV, const float* __restrict__ scale,
                                __bf16* __restrict__ Qn, __bf16* __restrict__ Knt,
                                __bf16* __restrict__ Vn) {
  int wave = threadIdx.x >> 5, lane = threadIdx.x & 31;
  int flat = blockIdx.x * 8 + wave;          // (b*NT + tok)*NHH + h
  int h = flat % NHH;
  int bt = flat / NHH;
  int tok = bt % NT;
  int b = bt / NT;
  const __bf16* base = QKV + (size_t)bt * QKVC + h * HD;
  float q[3], k[3], v[3];
  float sq = 0.f, sk = 0.f;
#pragma unroll
  for (int e = 0; e < 3; ++e) {
    int c = lane + 32 * e;
    q[e] = (float)base[c];
    k[e] = (float)base[DIMC + c];
    v[e] = (float)base[2 * DIMC + c];
    sq += q[e] * q[e];
    sk += k[e] * k[e];
  }
  for (int m = 16; m; m >>= 1) { sq += __shfl_xor(sq, m, 32); sk += __shfl_xor(sk, m, 32); }
  float sc = fminf(scale[h], 100.f);
  float iq = sc / fmaxf(sqrtf(sq), 1e-12f);
  float ik = 1.f / fmaxf(sqrtf(sk), 1e-12f);
  size_t bh = (size_t)b * NHH + h;
#pragma unroll
  for (int e = 0; e < 3; ++e) {
    int c = lane + 32 * e;
    Qn[(bh * NTP + tok) * HD + c]  = (__bf16)(q[e] * iq);
    Knt[(bh * HD + c) * NTP + tok] = (__bf16)(k[e] * ik);
    Vn[(bh * NTP + tok) * HD + c]  = (__bf16)v[e];
  }
}

// ---------------- dynamic position-bias MLP: 841 rows ----------------
__global__ void posbias_mlp_kernel(const float* __restrict__ biases,
    const float* pp_w, const float* pp_b,
    const float* ln1w, const float* ln1b, const float* f1w, const float* f1b,
    const float* ln2w, const float* ln2b, const float* f2w, const float* f2b,
    const float* ln3w, const float* ln3b, const float* f3w, const float* f3b,
    float* __restrict__ out) {
  int row = blockIdx.x, t = threadIdx.x;
  __shared__ float a[48], tmp[48], mv[2];
  if (t < 48) a[t] = biases[row * 2] * pp_w[t * 2] + biases[row * 2 + 1] * pp_w[t * 2 + 1] + pp_b[t];
  __syncthreads();
  const float* lnw[3] = {ln1w, ln2w, ln3w};
  const float* lnb[3] = {ln1b, ln2b, ln3b};
  const float* fw[2]  = {f1w, f2w};
  const float* fb[2]  = {f1b, f2b};
  for (int s = 0; s < 3; ++s) {
    if (t == 0) {
      float m = 0.f; for (int j = 0; j < 48; ++j) m += a[j]; m *= (1.f / 48.f);
      float v = 0.f; for (int j = 0; j < 48; ++j) { float d = a[j] - m; v += d * d; } v *= (1.f / 48.f);
      mv[0] = m; mv[1] = rsqrtf(v + 1e-5f);
    }
    __syncthreads();
    if (t < 48) tmp[t] = fmaxf((a[t] - mv[0]) * mv[1] * lnw[s][t] + lnb[s][t], 0.f);
    __syncthreads();
    if (s < 2) {
      if (t < 48) {
        float acc = fb[s][t];
        for (int j = 0; j < 48; ++j) acc += tmp[j] * fw[s][t * 48 + j];
        a[t] = acc;
      }
      __syncthreads();
    }
  }
  if (t < 8) {
    float acc = f3b[t];
    for (int j = 0; j < 48; ++j) acc += tmp[j] * f3w[t * 48 + j];
    out[row * 8 + t] = acc;
  }
}

// ---------------- batched attention scores: S(bh, NTP, NTP) = Qn @ Knt ----------------
__global__ void attn_score_kernel(const __bf16* __restrict__ Qn, const __bf16* __restrict__ Knt,
                                  float* __restrict__ S) {
  int wave = threadIdx.x >> 5, lane = threadIdx.x & 31;
  int tile = blockIdx.x * 8 + wave;          // 512 * 14 * 14 tiles
  int tj = tile % 14; tile /= 14;
  int ti = tile % 14;
  int bh = tile / 14;
  const __bf16* pA = Qn + ((size_t)bh * NTP + ti * 16) * HD
                     + (size_t)(lane & 15) * HD + ((lane >> 4) << 3);
  const __bf16* B = Knt + (size_t)bh * HD * NTP + tj * 16;
  v8f acc;
#pragma unroll
  for (int i = 0; i < 8; ++i) acc[i] = 0.f;
#pragma unroll
  for (int kb = 0; kb < HD; kb += 32) {
    v16bf a = load_a_ptr(pA); pA += 32;
    v16bf b = load_b_frag(B, NTP, kb, lane);
    acc = WMMA_BF16(a, b, acc);
  }
  int r = (lane >> 4) << 3, c = lane & 15;
  float* outp = S + ((size_t)bh * NTP + ti * 16) * NTP + tj * 16;
#pragma unroll
  for (int i2 = 0; i2 < 8; ++i2) outp[(size_t)(i2 + r) * NTP + c] = acc[i2];
}

// ---------------- pre-mix over heads + RPB + mask + softmax -> bf16 probs ----------------
__global__ void mix_mask_softmax_kernel(const float* __restrict__ S, const float* __restrict__ pb,
                                        const int* __restrict__ rel_idx,
                                        const float* __restrict__ pre_w,
                                        const float* __restrict__ prior_adj,
                                        __bf16* __restrict__ Ssm) {
  int flat = blockIdx.x;                     // (b*NHH + o)*NT + i
  int i = flat % NT;
  int bo = flat / NT;
  int o = bo % NHH;
  int b = bo / NHH;
  int j = threadIdx.x;
  __shared__ float sh[256];
  float val = -3.0e38f;
  if (j < NT) {
    float acc = 0.f;
    int ridx = rel_idx[i * NT + j];
#pragma unroll
    for (int h = 0; h < NHH; ++h) {
      float s = S[((size_t)(b * NHH + h) * NTP + i) * NTP + j];
      acc += pre_w[o * NHH + h] * (s + pb[ridx * NHH + h]);
    }
    if (!(prior_adj[((size_t)o * NT + i) * NT + j] > 0.f)) acc = -1e9f;
    val = acc;
  }
  sh[threadIdx.x] = val;
  __syncthreads();
  for (int st = 128; st > 0; st >>= 1) {
    if (threadIdx.x < st) sh[threadIdx.x] = fmaxf(sh[threadIdx.x], sh[threadIdx.x + st]);
    __syncthreads();
  }
  float mx = sh[0];
  __syncthreads();
  float e = (j < NT) ? __expf(val - mx) : 0.f;
  sh[threadIdx.x] = e;
  __syncthreads();
  for (int st = 128; st > 0; st >>= 1) {
    if (threadIdx.x < st) sh[threadIdx.x] += sh[threadIdx.x + st];
    __syncthreads();
  }
  float inv = 1.f / sh[0];
  if (j < NTP) Ssm[((size_t)bo * NTP + i) * NTP + j] = (__bf16)(e * inv);
}

// ---------------- post-mix over heads -> bf16 probs P (padded rows zeroed) ----------------
__global__ void postmix_kernel(const __bf16* __restrict__ Ssm, const float* __restrict__ post_w,
                               __bf16* __restrict__ P) {
  size_t idx = (size_t)blockIdx.x * 256 + threadIdx.x;   // 512*NTP*NTP exactly
  int j = (int)(idx % NTP);
  size_t t = idx / NTP;
  int i = (int)(t % NTP);
  size_t bo = t / NTP;
  int o = (int)(bo % NHH);
  size_t b = bo / NHH;
  float v = 0.f;
  if (i < NT) {
#pragma unroll
    for (int h = 0; h < NHH; ++h)
      v += post_w[o * NHH + h] * (float)Ssm[((b * NHH + h) * NTP + i) * NTP + j];
  }
  P[idx] = (__bf16)v;
}

// ---------------- batched P @ V -> Obf (MROWS x DIMC) bf16 ----------------
__global__ void attn_v_kernel(const __bf16* __restrict__ P, const __bf16* __restrict__ Vn,
                              __bf16* __restrict__ Obf) {
  int wave = threadIdx.x >> 5, lane = threadIdx.x & 31;
  int tile = blockIdx.x * 8 + wave;          // 512 * 14 * 6 tiles
  int tj = tile % 6; tile /= 6;
  int ti = tile % 14;
  int bh = tile / 14;
  int h = bh % NHH, b = bh / NHH;
  const __bf16* pA = P + ((size_t)bh * NTP + ti * 16) * NTP
                     + (size_t)(lane & 15) * NTP + ((lane >> 4) << 3);
  const __bf16* B = Vn + (size_t)bh * NTP * HD + tj * 16;
  v8f acc;
#pragma unroll
  for (int i = 0; i < 8; ++i) acc[i] = 0.f;
#pragma unroll
  for (int kb = 0; kb < NTP; kb += 32) {
    v16bf a = load_a_ptr(pA); pA += 32;
    v16bf bfr = load_b_frag(B, HD, kb, lane);
    acc = WMMA_BF16(a, bfr, acc);
  }
  int r = (lane >> 4) << 3, c = lane & 15;
#pragma unroll
  for (int i2 = 0; i2 < 8; ++i2) {
    int tok = ti * 16 + r + i2;
    if (tok < NT)
      Obf[((size_t)b * NT + tok) * DIMC + h * HD + tj * 16 + c] = (__bf16)acc[i2];
  }
}

// ---------------- workspace layout (bytes, all 256-aligned) ----------------
#define OFF_HN    ((size_t)0)              // 12608*768 bf16
#define OFF_WQKV  ((size_t)19365888)       // 768x2304 bf16 (K-major)
#define OFF_WPROJ ((size_t)22904832)       // 768x768
#define OFF_WM1   ((size_t)24084480)       // 768x3072
#define OFF_WM2   ((size_t)28803072)       // 3072x768
#define OFF_QKV   ((size_t)33521664)       // 12608x2304 bf16
#define OFF_QN    ((size_t)91619328)       // 512x224x96 bf16
#define OFF_KNT   ((size_t)113639424)      // 512x96x224 bf16
#define OFF_VN    ((size_t)135659520)      // 512x224x96 bf16
#define OFF_S     ((size_t)157679616)      // 512x224x224 fp32
#define OFF_SSM   ((size_t)260440064)      // 512x224x224 bf16
#define OFF_P     ((size_t)311820288)      // 512x224x224 bf16
#define OFF_PB    ((size_t)363200512)      // 841x8 fp32
#define OFF_OBF   ((size_t)363233280)      // 12608x768 bf16
#define OFF_M1    ((size_t)382599168)      // 12608x3072 bf16
// total ws need ~460.1 MB

extern "C" void kernel_launch(void* const* d_in, const int* in_sizes, int n_in,
                              void* d_out, int out_size, void* d_ws, size_t ws_size,
                              hipStream_t stream) {
  const float* x        = (const float*)d_in[0];
  const float* norm1_w  = (const float*)d_in[1];
  const float* norm1_b  = (const float*)d_in[2];
  const float* qkv_w    = (const float*)d_in[3];
  const float* scale    = (const float*)d_in[4];
  const float* pp_w     = (const float*)d_in[5];
  const float* pp_b     = (const float*)d_in[6];
  const float* ln1_w    = (const float*)d_in[7];
  const float* ln1_b    = (const float*)d_in[8];
  const float* fcp1_w   = (const float*)d_in[9];
  const float* fcp1_b   = (const float*)d_in[10];
  const float* ln2_w    = (const float*)d_in[11];
  const float* ln2_b    = (const float*)d_in[12];
  const float* fcp2_w   = (const float*)d_in[13];
  const float* fcp2_b   = (const float*)d_in[14];
  const float* ln3_w    = (const float*)d_in[15];
  const float* ln3_b    = (const float*)d_in[16];
  const float* fcp3_w   = (const float*)d_in[17];
  const float* fcp3_b   = (const float*)d_in[18];
  const float* pre_w    = (const float*)d_in[19];
  const float* post_w   = (const float*)d_in[20];
  const float* proj_w   = (const float*)d_in[21];
  const float* proj_b   = (const float*)d_in[22];
  const float* norm2_w  = (const float*)d_in[23];
  const float* norm2_b  = (const float*)d_in[24];
  const float* m1_w     = (const float*)d_in[25];
  const float* m1_b     = (const float*)d_in[26];
  const float* m2_w     = (const float*)d_in[27];
  const float* m2_b     = (const float*)d_in[28];
  const float* prior_adj= (const float*)d_in[29];
  const float* biases   = (const float*)d_in[30];
  const int*   rel_idx  = (const int*)d_in[31];
  float* out = (float*)d_out;
  char* ws = (char*)d_ws;

  __bf16* Hn   = (__bf16*)(ws + OFF_HN);
  __bf16* Wqkv = (__bf16*)(ws + OFF_WQKV);
  __bf16* Wproj= (__bf16*)(ws + OFF_WPROJ);
  __bf16* Wm1  = (__bf16*)(ws + OFF_WM1);
  __bf16* Wm2  = (__bf16*)(ws + OFF_WM2);
  __bf16* QKVb = (__bf16*)(ws + OFF_QKV);
  __bf16* Qn   = (__bf16*)(ws + OFF_QN);
  __bf16* Knt  = (__bf16*)(ws + OFF_KNT);
  __bf16* Vn   = (__bf16*)(ws + OFF_VN);
  float*  S    = (float*)(ws + OFF_S);
  __bf16* Ssm  = (__bf16*)(ws + OFF_SSM);
  __bf16* P    = (__bf16*)(ws + OFF_P);
  float*  Pb   = (float*)(ws + OFF_PB);
  __bf16* Obf  = (__bf16*)(ws + OFF_OBF);
  __bf16* M1b  = (__bf16*)(ws + OFF_M1);

  // weights -> bf16, K-major
  convT_kernel<<<(QKVC * DIMC + 255) / 256, 256, 0, stream>>>(qkv_w, Wqkv, QKVC, DIMC);
  convT_kernel<<<(DIMC * DIMC + 255) / 256, 256, 0, stream>>>(proj_w, Wproj, DIMC, DIMC);
  convT_kernel<<<(HIDC * DIMC + 255) / 256, 256, 0, stream>>>(m1_w, Wm1, HIDC, DIMC);
  convT_kernel<<<(DIMC * HIDC + 255) / 256, 256, 0, stream>>>(m2_w, Wm2, DIMC, HIDC);

  // LN1
  ln_bf16_kernel<<<MROWS, 256, 0, stream>>>(x, norm1_w, norm1_b, Hn);

  // QKV GEMM (no bias), bf16 out
  gemm_bf16_kernel<<<dim3(QKVC / 128, (MROWS + 127) / 128), 256, 0, stream>>>(
      Hn, Wqkv, nullptr, nullptr, nullptr, QKVb, MROWS, QKVC, DIMC, 0);

  // l2-normalize q,k; scale; layouts for WMMA
  qkv_post_kernel<<<MROWS, 256, 0, stream>>>(QKVb, scale, Qn, Knt, Vn);

  // position-bias MLP
  posbias_mlp_kernel<<<841, 64, 0, stream>>>(biases, pp_w, pp_b,
      ln1_w, ln1_b, fcp1_w, fcp1_b, ln2_w, ln2_b, fcp2_w, fcp2_b,
      ln3_w, ln3_b, fcp3_w, fcp3_b, Pb);

  // scores: 512 * 14 * 14 tiles, 8 tiles per block
  attn_score_kernel<<<(BB * NHH * 14 * 14) / 8, 256, 0, stream>>>(Qn, Knt, S);

  // pre-mix + rpb + mask + softmax
  mix_mask_softmax_kernel<<<BB * NHH * NT, 256, 0, stream>>>(S, Pb, rel_idx, pre_w, prior_adj, Ssm);

  // post-mix -> bf16 probs
  postmix_kernel<<<(BB * NHH * NTP * NTP) / 256, 256, 0, stream>>>(Ssm, post_w, P);

  // P @ V -> Obf
  attn_v_kernel<<<(BB * NHH * 14 * 6) / 8, 256, 0, stream>>>(P, Vn, Obf);

  // proj + residual -> d_out (fp32)
  gemm_bf16_kernel<<<dim3(DIMC / 128, (MROWS + 127) / 128), 256, 0, stream>>>(
      Obf, Wproj, proj_b, x, out, nullptr, MROWS, DIMC, DIMC, 2);

  // LN2 on d_out
  ln_bf16_kernel<<<MROWS, 256, 0, stream>>>(out, norm2_w, norm2_b, Hn);

  // MLP1 + exact GELU -> bf16
  gemm_bf16_kernel<<<dim3(HIDC / 128, (MROWS + 127) / 128), 256, 0, stream>>>(
      Hn, Wm1, m1_b, nullptr, nullptr, M1b, MROWS, HIDC, DIMC, 1);

  // MLP2, residual add in place into d_out
  gemm_bf16_kernel<<<dim3(DIMC / 128, (MROWS + 127) / 128), 256, 0, stream>>>(
      M1b, Wm2, m2_b, nullptr, out, nullptr, MROWS, DIMC, HIDC, 3);
}
